// Multihead_Attention_18073222382259
// MI455X (gfx1250) — matrix operand order repaired and checked by reading, not compiled, not run on hardware
//
#include <hip/hip_runtime.h>

typedef __attribute__((ext_vector_type(8)))  __bf16 v8bf;
typedef __attribute__((ext_vector_type(16))) __bf16 v16bf;
typedef __attribute__((ext_vector_type(8)))  float  v8f;

#define H 12
#define HD 64
#define NSEQ 1024
#define BATCH 8
#define DMODEL 768
#define QKVCOLS 2304
#define ROWS (BATCH * NSEQ)   // 8192

__device__ __forceinline__ unsigned short cvt_bf16(float f) {
    unsigned int u = __float_as_uint(f);
    u += 0x7FFFu + ((u >> 16) & 1u);   // round-to-nearest-even
    return (unsigned short)(u >> 16);
}

// A-matrix fragment 16x32 bf16 (ISA 7.12.2): lane lo = row, chunks at k=hi*8 and 16+hi*8
__device__ __forceinline__ v16bf load_a16(const unsigned short* base, int ld) {
    int lane = threadIdx.x & 31;
    int lo = lane & 15, hi = lane >> 4;
    const unsigned short* p = base + lo * ld + hi * 8;
    v8bf a0 = *(const v8bf*)(p);
    v8bf a1 = *(const v8bf*)(p + 16);
    v16bf r;
#pragma unroll
    for (int i = 0; i < 8; ++i) { r[i] = a0[i]; r[i + 8] = a1[i]; }
    return r;
}

// B-matrix fragment 32x16 bf16, source stored transposed [n][k]: lane lo = n, chunks at k=hi*8 and 16+hi*8
__device__ __forceinline__ v16bf load_b16(const unsigned short* base, int ld) {
    int lane = threadIdx.x & 31;
    int lo = lane & 15, hi = lane >> 4;
    const unsigned short* p = base + lo * ld + hi * 8;
    v8bf a0 = *(const v8bf*)(p);
    v8bf a1 = *(const v8bf*)(p + 16);
    v16bf r;
#pragma unroll
    for (int i = 0; i < 8; ++i) { r[i] = a0[i]; r[i + 8] = a1[i]; }
    return r;
}

__device__ __forceinline__ v8f zero8() {
    v8f z = {0.f, 0.f, 0.f, 0.f, 0.f, 0.f, 0.f, 0.f};
    return z;
}

#define WMMA_BF16(A, B, C) \
    __builtin_amdgcn_wmma_f32_16x16x32_bf16(false, (A), false, (B), (short)0, (C), false, false)

// ---------------- conversion kernels ----------------
__global__ void k_cvt(const float* __restrict__ in, unsigned short* __restrict__ out, int n) {
    int i = blockIdx.x * blockDim.x + threadIdx.x;
    if (i < n) out[i] = cvt_bf16(in[i]);
}

// out[c][r] = in[r][c]  (weights [in,out] -> bf16 [out][in])
__global__ void k_cvt_t(const float* __restrict__ in, unsigned short* __restrict__ out,
                        int rows_in, int cols_in) {
    int i = blockIdx.x * blockDim.x + threadIdx.x;
    int n = rows_in * cols_in;
    if (i < n) {
        int r = i % rows_in;
        int c = i / rows_in;
        out[(size_t)c * rows_in + r] = cvt_bf16(in[(size_t)r * cols_in + c]);
    }
}

// 32x64 tile per wave, two-stage software pipeline with no buffer-rotation copies.
// Last half-iteration reloads offset 0 as a discarded dummy (branchless, in-bounds).
#define GEMM_K_LOOP(A0_, A1_, B0_)                                                     \
    v16bf a0A = load_a16((A0_), DMODEL);                                               \
    v16bf a1A = load_a16((A1_), DMODEL);                                               \
    v16bf bA[4];                                                                       \
    _Pragma("unroll")                                                                  \
    for (int t = 0; t < 4; ++t) bA[t] = load_b16((B0_) + (size_t)t * 16 * DMODEL, DMODEL); \
    for (int k0 = 0; k0 < DMODEL; k0 += 64) {                                          \
        int k1 = k0 + 32;                                                              \
        __builtin_prefetch((A0_) + k1 + 32, 0, 1);                                     \
        __builtin_prefetch((B0_) + k1 + 32, 0, 1);                                     \
        v16bf a0B = load_a16((A0_) + k1, DMODEL);                                      \
        v16bf a1B = load_a16((A1_) + k1, DMODEL);                                      \
        v16bf bB[4];                                                                   \
        _Pragma("unroll")                                                              \
        for (int t = 0; t < 4; ++t) bB[t] = load_b16((B0_) + (size_t)t * 16 * DMODEL + k1, DMODEL); \
        _Pragma("unroll")                                                              \
        for (int t = 0; t < 4; ++t) acc[0][t] = WMMA_BF16(a0A, bA[t], acc[0][t]);      \
        _Pragma("unroll")                                                              \
        for (int t = 0; t < 4; ++t) acc[1][t] = WMMA_BF16(a1A, bA[t], acc[1][t]);      \
        int k2 = (k0 + 64 < DMODEL) ? (k0 + 64) : 0;                                   \
        a0A = load_a16((A0_) + k2, DMODEL);                                            \
        a1A = load_a16((A1_) + k2, DMODEL);                                            \
        _Pragma("unroll")                                                              \
        for (int t = 0; t < 4; ++t) bA[t] = load_b16((B0_) + (size_t)t * 16 * DMODEL + k2, DMODEL); \
        _Pragma("unroll")                                                              \
        for (int t = 0; t < 4; ++t) acc[0][t] = WMMA_BF16(a0B, bB[t], acc[0][t]);      \
        _Pragma("unroll")                                                              \
        for (int t = 0; t < 4; ++t) acc[1][t] = WMMA_BF16(a1B, bB[t], acc[1][t]);      \
    }

// ---------------- QKV GEMM: [8192,768]bf16 @ wT[2304,768]bf16 -> q/k/vt ----------------
__global__ __launch_bounds__(128) void k_qkv(const unsigned short* __restrict__ xb,
                                             const unsigned short* __restrict__ wt,
                                             unsigned short* __restrict__ q,
                                             unsigned short* __restrict__ kk,
                                             unsigned short* __restrict__ vt) {
    int wave = threadIdx.x >> 5;
    int wid = blockIdx.x * 4 + wave;      // 0 .. 256*36-1
    int m32 = wid % (ROWS / 32);
    int nb  = wid / (ROWS / 32);          // 0..35: which*12 + head
    int rowbase = m32 * 32;
    int colbase = nb * 64;
    int lane = threadIdx.x & 31, lo = lane & 15, hi = lane >> 4;

    const unsigned short* A0 = xb + (size_t)rowbase * DMODEL;
    const unsigned short* A1 = A0 + 16 * DMODEL;
    const unsigned short* B0 = wt + (size_t)colbase * DMODEL;

    v8f acc[2][4];
#pragma unroll
    for (int g = 0; g < 2; ++g)
#pragma unroll
        for (int t = 0; t < 4; ++t) acc[g][t] = zero8();

    GEMM_K_LOOP(A0, A1, B0)

    int which = nb / H, h = nb % H;
#pragma unroll
    for (int g = 0; g < 2; ++g) {
#pragma unroll
        for (int t = 0; t < 4; ++t) {
#pragma unroll
            for (int v = 0; v < 8; ++v) {
                int grow = rowbase + g * 16 + hi * 8 + v;
                int bb = grow >> 10, n = grow & 1023;
                int d = t * 16 + lo;
                unsigned short val = cvt_bf16(acc[g][t][v]);
                if (which == 0)
                    q[((size_t)(bb * H + h) * NSEQ + n) * HD + d] = val;
                else if (which == 1)
                    kk[((size_t)(bb * H + h) * NSEQ + n) * HD + d] = val;
                else
                    vt[((size_t)(bb * H + h) * HD + d) * NSEQ + n] = val;
            }
        }
    }
}

// ---------------- flash attention: per wave, 16 query rows of one (b,h) ----------------
__global__ __launch_bounds__(128) void k_attn(const unsigned short* __restrict__ q,
                                              const unsigned short* __restrict__ kk,
                                              const unsigned short* __restrict__ vt,
                                              unsigned short* __restrict__ attn) {
    __shared__ __align__(16) unsigned short pst[4][16 * 32];
    int wave = threadIdx.x >> 5;
    int wid = blockIdx.x * 4 + wave;      // 0..6143
    int bh = wid >> 6;                    // batch*head
    int it = wid & 63;                    // query tile
    int lane = threadIdx.x & 31, lo = lane & 15, hi = lane >> 4;

    const unsigned short* Qb = q + ((size_t)bh * NSEQ + it * 16) * HD;
    const unsigned short* Kb = kk + (size_t)bh * NSEQ * HD;
    const unsigned short* Vb = vt + (size_t)bh * HD * NSEQ;

    v16bf qa0 = load_a16(Qb, HD);
    v16bf qa1 = load_a16(Qb + 32, HD);

    float mrow[8], lrow[8];
#pragma unroll
    for (int v = 0; v < 8; ++v) { mrow[v] = -1e30f; lrow[v] = 0.f; }
    v8f o[4];
#pragma unroll
    for (int t = 0; t < 4; ++t) o[t] = zero8();

    // K fragments double-buffered across 16-column sub-tiles
    v16bf kc0 = load_b16(Kb, HD);
    v16bf kc1 = load_b16(Kb + 32, HD);

#pragma unroll 2
    for (int j0 = 0; j0 < NSEQ; j0 += 32) {
        // preload V fragments for this 32-column block; consumed after the retile barrier
        v16bf vbf[4];
#pragma unroll
        for (int t = 0; t < 4; ++t)
            vbf[t] = load_b16(Vb + (size_t)(t * 16) * NSEQ + j0, NSEQ);

#pragma unroll
        for (int sub = 0; sub < 2; ++sub) {
            v8f s = zero8();
            s = WMMA_BF16(qa0, kc0, s);
            s = WMMA_BF16(qa1, kc1, s);
            // prefetch next sub-tile's K fragments (wraps harmlessly on last iter)
            int jn = (j0 + (sub + 1) * 16) & (NSEQ - 1);
            v16bf kn0 = load_b16(Kb + (size_t)jn * HD, HD);
            v16bf kn1 = load_b16(Kb + (size_t)jn * HD + 32, HD);
#pragma unroll
            for (int v = 0; v < 8; ++v) {
                float sv = s[v];
                float t = sv;
                t = fmaxf(t, __shfl_xor(t, 1, 32));
                t = fmaxf(t, __shfl_xor(t, 2, 32));
                t = fmaxf(t, __shfl_xor(t, 4, 32));
                t = fmaxf(t, __shfl_xor(t, 8, 32));
                float mn = fmaxf(mrow[v], t);
                float alpha = __expf(mrow[v] - mn);
                float p = __expf(sv - mn);
                float ps = p;
                ps += __shfl_xor(ps, 1, 32);
                ps += __shfl_xor(ps, 2, 32);
                ps += __shfl_xor(ps, 4, 32);
                ps += __shfl_xor(ps, 8, 32);
                lrow[v] = lrow[v] * alpha + ps;
                mrow[v] = mn;
#pragma unroll
                for (int t2 = 0; t2 < 4; ++t2) o[t2][v] *= alpha;
                pst[wave][(hi * 8 + v) * 32 + sub * 16 + lo] = cvt_bf16(p);
            }
            kc0 = kn0; kc1 = kn1;
        }
        __syncthreads();
        v16bf pa = load_a16(&pst[wave][0], 32);
#pragma unroll
        for (int t = 0; t < 4; ++t)
            o[t] = WMMA_BF16(pa, vbf[t], o[t]);
        __syncthreads();
    }

    // reference quirk: att = softmax(energy) / SCALE, SCALE = HD^-0.5 = 0.125 -> x8
    int b = bh / H, h = bh % H;
#pragma unroll
    for (int v = 0; v < 8; ++v) {
        float inv = 8.0f / lrow[v];
        int i = it * 16 + hi * 8 + v;
#pragma unroll
        for (int t = 0; t < 4; ++t) {
            attn[((size_t)(b * NSEQ + i)) * DMODEL + h * HD + t * 16 + lo] =
                cvt_bf16(o[t][v] * inv);
        }
    }
}

// ---------------- projection: 32x64 tile per wave, out = attn @ w_proj + b ----------------
__global__ __launch_bounds__(128) void k_proj(const unsigned short* __restrict__ attn,
                                              const unsigned short* __restrict__ wt,
                                              const float* __restrict__ bias,
                                              float* __restrict__ out) {
    int wave = threadIdx.x >> 5;
    int wid = blockIdx.x * 4 + wave;      // 0 .. 256*12-1
    int m32 = wid % (ROWS / 32);
    int nb  = wid / (ROWS / 32);          // 0..11
    int rowbase = m32 * 32;
    int colbase = nb * 64;
    int lane = threadIdx.x & 31, lo = lane & 15, hi = lane >> 4;

    const unsigned short* A0 = attn + (size_t)rowbase * DMODEL;
    const unsigned short* A1 = A0 + 16 * DMODEL;
    const unsigned short* B0 = wt + (size_t)colbase * DMODEL;

    v8f acc[2][4];
#pragma unroll
    for (int t = 0; t < 4; ++t) {
        float bv = bias[colbase + t * 16 + lo];
#pragma unroll
        for (int v = 0; v < 8; ++v) { acc[0][t][v] = bv; acc[1][t][v] = bv; }
    }

    GEMM_K_LOOP(A0, A1, B0)

#pragma unroll
    for (int g = 0; g < 2; ++g) {
#pragma unroll
        for (int t = 0; t < 4; ++t) {
#pragma unroll
            for (int v = 0; v < 8; ++v) {
                int grow = rowbase + g * 16 + hi * 8 + v;
                out[(size_t)grow * DMODEL + colbase + t * 16 + lo] = acc[g][t][v];
            }
        }
    }
}

extern "C" void kernel_launch(void* const* d_in, const int* in_sizes, int n_in,
                              void* d_out, int out_size, void* d_ws, size_t ws_size,
                              hipStream_t stream) {
    const float* x      = (const float*)d_in[0];   // [8,1024,768]
    const float* w_qkv  = (const float*)d_in[1];   // [768,2304]
    const float* w_proj = (const float*)d_in[2];   // [768,768]
    const float* b_proj = (const float*)d_in[3];   // [768]
    float* out = (float*)d_out;

    char* ws = (char*)d_ws;
    size_t off = 0;
    unsigned short* xb     = (unsigned short*)(ws + off); off += (size_t)ROWS * DMODEL * 2;     // 12.58 MB
    unsigned short* wqkvT  = (unsigned short*)(ws + off); off += (size_t)QKVCOLS * DMODEL * 2;  // 3.54 MB
    unsigned short* wprojT = (unsigned short*)(ws + off); off += (size_t)DMODEL * DMODEL * 2;   // 1.18 MB
    unsigned short* qbuf   = (unsigned short*)(ws + off); off += (size_t)ROWS * DMODEL * 2;
    unsigned short* kbuf   = (unsigned short*)(ws + off); off += (size_t)ROWS * DMODEL * 2;
    unsigned short* vtbuf  = (unsigned short*)(ws + off); off += (size_t)ROWS * DMODEL * 2;
    unsigned short* attnb  = (unsigned short*)(ws + off); off += (size_t)ROWS * DMODEL * 2;

    // convert inputs to bf16 (weights transposed to [out][in])
    {
        int n = ROWS * DMODEL;
        k_cvt<<<(n + 255) / 256, 256, 0, stream>>>(x, xb, n);
    }
    {
        int n = DMODEL * QKVCOLS;
        k_cvt_t<<<(n + 255) / 256, 256, 0, stream>>>(w_qkv, wqkvT, DMODEL, QKVCOLS);
    }
    {
        int n = DMODEL * DMODEL;
        k_cvt_t<<<(n + 255) / 256, 256, 0, stream>>>(w_proj, wprojT, DMODEL, DMODEL);
    }

    // QKV GEMM: 256 m-tiles * 36 col-blocks = 9216 waves -> 2304 blocks of 128
    k_qkv<<<(ROWS / 32) * 36 / 4, 128, 0, stream>>>(xb, wqkvT, qbuf, kbuf, vtbuf);

    // attention: 8*12*64 = 6144 waves -> 1536 blocks of 128
    k_attn<<<BATCH * H * (NSEQ / 16) / 4, 128, 0, stream>>>(qbuf, kbuf, vtbuf, attnb);

    // projection: 256 * 12 = 3072 waves -> 768 blocks of 128
    k_proj<<<(ROWS / 32) * 12 / 4, 128, 0, stream>>>(attnb, wprojT, b_proj, out);
}
